// MinCostMaxFlow_8495445311587
// MI455X (gfx1250) — matrix operand order repaired and checked by reading, not compiled, not run on hardware
//
#include <hip/hip_runtime.h>
#include <math.h>

#define N_CLASSES 19
#define UNI       256
#define N_POINTS  12544
#define TOPK      5
#define HWD       512
#define HW2       (512*512)
#define N_NODES   (N_CLASSES + UNI + 2)       // 277
#define S_NODE    0
#define T_NODE    (N_NODES - 1)               // 276
#define N_EDGES   (2*TOPK*N_CLASSES + 2*UNI + 2*N_CLASSES)  // 740
#define E_COST    (2*TOPK*N_CLASSES)          // 190
#define KSPLIT    8
#define KCHUNK    (N_POINTS / KSPLIT)         // 1568 = 49*32

typedef __attribute__((ext_vector_type(16))) _Float16 v16h;
typedef __attribute__((ext_vector_type(8)))  _Float16 v8h;
typedef __attribute__((ext_vector_type(8)))  float    v8f;

// ---- workspace layout (bytes) ----
#define WS_A     0                        // 256 f32       (softplus sums)
#define WS_PRES  1024                     // 32 int        (class-present flags)
#define WS_M     4096                     // 32*256 f32    (Tm @ x^T, padded M)
#define WS_TM    65536                    // 32*12544 f16  (padded one-hot Tm)
#define WS_XT    868352                   // 256*12544 f16 (softmax probs, point-major)

// ---------------------------------------------------------------------------
__global__ __launch_bounds__(256) void mcmf_init_kernel(float* __restrict__ A,
                                                        int* __restrict__ pres,
                                                        float* __restrict__ M)
{
    int t = threadIdx.x;
    A[t] = 0.0f;
    if (t < 32) pres[t] = 0;
    for (int i = t; i < 32 * UNI; i += 256) M[i] = 0.0f;
}

// ---------------------------------------------------------------------------
// One wave per point (8 channels per lane). 392 blocks * 8 waves * 4 points.
__global__ __launch_bounds__(256) void mcmf_sample_kernel(
    const float* __restrict__ logits, const int* __restrict__ target,
    const float* __restrict__ pc, float* __restrict__ A_sum,
    int* __restrict__ pres, _Float16* __restrict__ Tm, _Float16* __restrict__ XT)
{
    __shared__ float A_part[UNI];
    int tid  = threadIdx.x;
    int lane = tid & 31;
    int wave = tid >> 5;
    A_part[tid] = 0.0f;
    __syncthreads();

    float accA[8];
#pragma unroll
    for (int j = 0; j < 8; ++j) accA[j] = 0.0f;

    for (int k = 0; k < 4; ++k) {
        int p = blockIdx.x * 32 + wave * 4 + k;
        float px = pc[2 * p], py = pc[2 * p + 1];
        float ix = fminf(fmaxf(px * 512.0f - 0.5f, 0.0f), 511.0f);
        float iy = fminf(fmaxf(py * 512.0f - 0.5f, 0.0f), 511.0f);
        float x0f = floorf(ix), y0f = floorf(iy);
        float wx = ix - x0f, wy = iy - y0f;
        int x0 = (int)x0f, y0 = (int)y0f;
        int x1 = min(x0 + 1, 511), y1 = min(y0 + 1, 511);
        int xn = (int)rintf(ix), yn = (int)rintf(iy);   // round-half-even = jnp.round
        int tgt = target[yn * HWD + xn];
        float w00 = (1.0f - wx) * (1.0f - wy), w01 = wx * (1.0f - wy);
        float w10 = (1.0f - wx) * wy,          w11 = wx * wy;

        float v[8];
#pragma unroll
        for (int j = 0; j < 8; ++j) {
            int c = lane + 32 * j;
            const float* b = logits + (size_t)c * HW2;
            v[j] = b[y0 * HWD + x0] * w00 + b[y0 * HWD + x1] * w01 +
                   b[y1 * HWD + x0] * w10 + b[y1 * HWD + x1] * w11;
        }
        // softmax over 256 channels (8 per lane, wave32 reduction)
        float m = v[0];
#pragma unroll
        for (int j = 1; j < 8; ++j) m = fmaxf(m, v[j]);
        for (int off = 16; off > 0; off >>= 1) m = fmaxf(m, __shfl_xor(m, off, 32));
        float s = 0.0f;
#pragma unroll
        for (int j = 0; j < 8; ++j) { v[j] = expf(v[j] - m); s += v[j]; }
        for (int off = 16; off > 0; off >>= 1) s += __shfl_xor(s, off, 32);
        float inv = 1.0f / s;
#pragma unroll
        for (int j = 0; j < 8; ++j) {
            float x = v[j] * inv;
            int c = lane + 32 * j;
            XT[(size_t)c * N_POINTS + p] = (_Float16)x;      // point-major rows
            accA[j] += log1pf(expf(x));                       // softplus
        }
        Tm[(size_t)lane * N_POINTS + p] =
            (lane == tgt) ? (_Float16)1.0f : (_Float16)0.0f; // rows 19..31 zero
        if (lane == 0) pres[tgt] = 1;                         // benign race
    }
#pragma unroll
    for (int j = 0; j < 8; ++j) atomicAdd(&A_part[lane + 32 * j], accA[j]);
    __syncthreads();
    atomicAdd(&A_sum[tid], A_part[tid]);
}

// ---------------------------------------------------------------------------
// M = Tm(32 x 12544, f16) @ XT^T(12544 x 256, f16), f32 accumulate.
// One wave per 16x16 tile; K split across blockIdx.z with atomic reduction.
// Operands (A 16x32, B 32x16) use contiguous K order, half h: K = 16h..16h+15, same for A and B.
__global__ __launch_bounds__(32) void mcmf_wmma_kernel(
    const _Float16* __restrict__ Tm, const _Float16* __restrict__ XT,
    float* __restrict__ M)
{
    int lane   = threadIdx.x;
    int n_tile = blockIdx.x;   // 0..15
    int m_tile = blockIdx.y;   // 0..1
    int ks     = blockIdx.z;   // 0..KSPLIT-1
    int mn = lane & 15;        // A: row M ; B: col N
    int h  = lane >> 4;        // lane half
    const _Float16* arow = Tm + (size_t)(m_tile * 16 + mn) * N_POINTS;
    const _Float16* brow = XT + (size_t)(n_tile * 16 + mn) * N_POINTS;
    v8f acc = {};
    int kbeg = ks * KCHUNK, kend = kbeg + KCHUNK;
    for (int k0 = kbeg; k0 < kend; k0 += 32) {
        // A 16x32: lane half h covers K = h*16 .. h*16+15 (contiguous in Tm row)
        v16h a = *(const v16h*)(arow + k0 + h * 16);
        // B 32x16: lane half h covers K = h*16 .. h*16+15 (contiguous in XT row)
        v16h b = *(const v16h*)(brow + k0 + h * 16);
        acc = __builtin_amdgcn_wmma_f32_16x16x32_f16(
                  false, a, false, b, (short)0, acc, false, false);
    }
    // C/D layout: VGPR r -> M = r + 8*h, N = lane&15
#pragma unroll
    for (int r = 0; r < 8; ++r) {
        int Mr = m_tile * 16 + r + 8 * h;
        int Nc = n_tile * 16 + mn;
        atomicAdd(&M[Mr * UNI + Nc], acc[r]);
    }
}

// ---------------------------------------------------------------------------
// order-preserving float <-> uint keys (monotone, handles negatives and +inf)
__device__ __forceinline__ unsigned fkey(float f) {
    unsigned u = __float_as_uint(f);
    return (u & 0x80000000u) ? ~u : (u | 0x80000000u);
}
__device__ __forceinline__ float funkey(unsigned k) {
    unsigned u = (k & 0x80000000u) ? (k & 0x7fffffffu) : ~k;
    return __uint_as_float(u);
}

// Fused: cost matrix -> top-5 -> edge build -> successive-shortest-path MCMF
// -> output extraction. Single block, everything in LDS.
__global__ __launch_bounds__(256) void mcmf_solve_kernel(
    const float* __restrict__ bipart, const float* __restrict__ A_sum,
    const int* __restrict__ presG, const float* __restrict__ Mmat,
    float* __restrict__ out)
{
    __shared__ float cost[N_CLASSES][UNI];
    __shared__ int   pres[N_CLASSES];
    __shared__ int   ef[N_EDGES], et[N_EDGES];
    __shared__ float cap[N_EDGES], ecst[N_EDGES];
    __shared__ unsigned diskey[N_NODES];
    __shared__ float dis_f[N_NODES];
    __shared__ unsigned hkey[N_NODES];
    __shared__ float h_f[N_NODES];
    __shared__ int   pred[N_NODES];
    __shared__ unsigned char tightf[N_EDGES];
    __shared__ int   out_i[UNI];
    __shared__ int   changed, reachF;

    int tid = threadIdx.x;
    const float invN = 1.0f / (float)N_POINTS;
    const float BIGF = 3.0e38f;

    if (tid < N_CLASSES) pres[tid] = presG[tid];
    float Au = A_sum[tid] * invN;
    for (int c = 0; c < N_CLASSES; ++c)
        cost[c][tid] = 1.0f - bipart[c * UNI + tid] + Au - Mmat[c * UNI + tid] * invN;
    out_i[tid] = 255;
    __syncthreads();

    // top-5 per class (strict < scan matches lax.top_k tie-breaking) + edges
    if (tid < N_CLASSES) {
        int c = tid;
        float capf = pres[c] ? 1.0f : 0.0f;
        for (int r = 0; r < TOPK; ++r) {
            float best = cost[c][0]; int bu = 0;
            for (int u = 1; u < UNI; ++u) {
                float cv = cost[c][u];
                if (cv < best) { best = cv; bu = u; }
            }
            cost[c][bu] = INFINITY;
            int e = 2 * (c * TOPK + r);
            ef[e] = 1 + bu;        et[e] = 1 + UNI + c; cap[e] = capf; ecst[e] = best;
            ef[e+1] = 1 + UNI + c; et[e+1] = 1 + bu;    cap[e+1] = 0.0f; ecst[e+1] = -best;
        }
        int e = E_COST + 2 * UNI + 2 * c;
        ef[e] = 1 + UNI + c; et[e] = T_NODE;      cap[e] = 2.0f;  ecst[e] = 0.0f;
        ef[e+1] = T_NODE;    et[e+1] = 1 + UNI + c; cap[e+1] = 0.0f; ecst[e+1] = 0.0f;
    }
    {   // source <-> uni edges
        int u = tid, e = E_COST + 2 * u;
        ef[e] = S_NODE;  et[e] = 1 + u;  cap[e] = 1.0f; ecst[e] = 0.0f;
        ef[e+1] = 1 + u; et[e+1] = S_NODE; cap[e+1] = 0.0f; ecst[e+1] = 0.0f;
    }
    __syncthreads();

    for (int iter = 0; iter < 400; ++iter) {
        // ---- Bellman-Ford to fixed point (early exit on convergence) ----
        for (int v = tid; v < N_NODES; v += 256)
            diskey[v] = fkey(v == S_NODE ? 0.0f : INFINITY);
        if (tid == 0) changed = 1;
        __syncthreads();
        for (int round = 0; round < N_NODES; ++round) {
            if (!changed) break;
            __syncthreads();
            if (tid == 0) changed = 0;
            __syncthreads();
            for (int e = tid; e < N_EDGES; e += 256) {
                if (cap[e] > 0.0f) {
                    float cand = funkey(diskey[ef[e]]) + ecst[e];
                    if (cand < BIGF) {
                        unsigned ck = fkey(cand);
                        unsigned old = atomicMin(&diskey[et[e]], ck);
                        if (ck < old) changed = 1;
                    }
                }
            }
            __syncthreads();
        }
        for (int v = tid; v < N_NODES; v += 256) dis_f[v] = funkey(diskey[v]);
        __syncthreads();
        if (tid == 0) reachF = (dis_f[T_NODE] < BIGF) ? 1 : 0;
        __syncthreads();
        if (!reachF) break;

        // ---- tight edges ----
        for (int e = tid; e < N_EDGES; e += 256) {
            float dt = dis_f[et[e]];
            tightf[e] = (cap[e] > 0.0f) && (dis_f[ef[e]] + ecst[e] == dt) && (dt < BIGF);
        }
        // ---- hop-count BFS over tight edges ----
        for (int v = tid; v < N_NODES; v += 256)
            hkey[v] = fkey(v == S_NODE ? 0.0f : 1.0e9f);
        if (tid == 0) changed = 1;
        __syncthreads();
        for (int round = 0; round < N_NODES; ++round) {
            if (!changed) break;
            __syncthreads();
            if (tid == 0) changed = 0;
            __syncthreads();
            for (int e = tid; e < N_EDGES; e += 256) {
                if (tightf[e]) {
                    unsigned ck = fkey(funkey(hkey[ef[e]]) + 1.0f);
                    unsigned old = atomicMin(&hkey[et[e]], ck);
                    if (ck < old) changed = 1;
                }
            }
            __syncthreads();
        }
        for (int v = tid; v < N_NODES; v += 256) { h_f[v] = funkey(hkey[v]); pred[v] = N_EDGES; }
        __syncthreads();
        for (int e = tid; e < N_EDGES; e += 256)
            if (tightf[e] && (h_f[ef[e]] + 1.0f == h_f[et[e]]))
                atomicMin(&pred[et[e]], e);
        __syncthreads();

        // ---- serial augment along min-edge-id predecessor chain ----
        if (tid == 0) {
            int v = T_NODE, guard = 0; float f = INFINITY;
            while (v != S_NODE && guard++ < 1000) {
                int e = pred[v]; if (e >= N_EDGES) break;
                f = fminf(f, cap[e]); v = ef[e];
            }
            v = T_NODE; guard = 0;
            while (v != S_NODE && guard++ < 1000) {
                int e = pred[v]; if (e >= N_EDGES) break;
                cap[e] -= f; cap[e ^ 1] += f; v = ef[e];
            }
        }
        __syncthreads();
    }
    __syncthreads();

    // ---- output: per-uni min assigned class over saturated fwd edges ----
    for (int j = tid; j < TOPK * N_CLASSES; j += 256) {
        int e = 2 * j;
        int c = et[e] - 1 - UNI;
        int u = ef[e] - 1;
        if (pres[c] && cap[e] == 0.0f) atomicMin(&out_i[u], c);
    }
    __syncthreads();
    out[tid] = (float)out_i[tid];
}

// ---------------------------------------------------------------------------
extern "C" void kernel_launch(void* const* d_in, const int* in_sizes, int n_in,
                              void* d_out, int out_size, void* d_ws, size_t ws_size,
                              hipStream_t stream)
{
    const float* logits = (const float*)d_in[0];   // (1,256,512,512) f32
    const int*   target = (const int*)d_in[1];     // (1,512,512) i32
    const float* bipart = (const float*)d_in[2];   // (19,256) f32
    const float* pc     = (const float*)d_in[3];   // (1,12544,2) f32
    float* out = (float*)d_out;                    // 256 f32

    char* ws = (char*)d_ws;
    float*     A    = (float*)(ws + WS_A);
    int*       pres = (int*)(ws + WS_PRES);
    float*     M    = (float*)(ws + WS_M);
    _Float16*  Tm   = (_Float16*)(ws + WS_TM);
    _Float16*  XT   = (_Float16*)(ws + WS_XT);

    mcmf_init_kernel<<<1, 256, 0, stream>>>(A, pres, M);
    mcmf_sample_kernel<<<N_POINTS / 32, 256, 0, stream>>>(logits, target, pc,
                                                          A, pres, Tm, XT);
    mcmf_wmma_kernel<<<dim3(16, 2, KSPLIT), 32, 0, stream>>>(Tm, XT, M);
    mcmf_solve_kernel<<<1, 256, 0, stream>>>(bipart, A, pres, M, out);
}
